// circle_fine_grained_extractor_47201690583538
// MI455X (gfx1250) — compile-verified, matrix operand order checked by reading, not run on hardware
//
#include <hip/hip_runtime.h>
#include <hip/hip_bf16.h>

typedef unsigned short u16;
typedef __attribute__((ext_vector_type(16))) __bf16 v16bf;
typedef __attribute__((ext_vector_type(8)))  float  v8f;
typedef __attribute__((address_space(3))) u16 lds_u16;

#define SCALE_F 0.022097086912079612f  // 2048^-0.5

__device__ __forceinline__ u16 f2bf(float f) {
  union { float f; unsigned u; } c; c.f = f;
  unsigned r = c.u + 0x7FFFu + ((c.u >> 16) & 1u);
  return (u16)(r >> 16);
}

union FragBF { v16bf v; u16 u[16]; };

// CDNA5 async memory->LDS copy (16B per lane), tracked with ASYNCcnt.
__device__ __forceinline__ void async_ld16(u16* dst_lds, const void* src_global) {
  lds_u16* l = (lds_u16*)dst_lds;
  asm volatile("global_load_async_to_lds_b128 %0, %1, off"
               :: "v"(l), "v"(src_global) : "memory");
}
__device__ __forceinline__ void wait_async() {
  asm volatile("s_wait_asynccnt 0x0" ::: "memory");
}

// ---------------------------------------------------------------------------
// Batched bf16 WMMA GEMM:  C[z] = alpha * A[z] @ op(B[z]) (+ Cadd[z])
//   A   : bf16 [M,K] row-major
//   B   : bIsKN==0 -> [N,K] row-major (computes A @ B^T)
//         bIsKN==1 -> [K,N] row-major (computes A @ B, transposed while staging)
//   bF32: B elements are f32, converted to bf16 during LDS staging
//   bmode==1: per-batch B offset = group-of-4 mapping for f_cor (contig gws)
//   C (f32), Cbf (bf16), Cadd (f32) all nullable; same [M,N]/strideC layout.
// Block: 256 threads = 8 wave32, 64x64 tile, K-step 32, 2 wmma/acc per step.
// Double-buffered LDS; full tiles staged with GLOBAL_LOAD_ASYNC_TO_LDS_B128
// overlapping the WMMAs; one s_wait_asynccnt + one barrier per K-step.
// ---------------------------------------------------------------------------
#define TM 64
#define TN 64
#define TK 32
#define LDT 40

__global__ __launch_bounds__(256) void gemm_bf16(
    const u16* __restrict__ A, const void* __restrict__ Bm,
    float* __restrict__ C, u16* __restrict__ Cbf, const float* __restrict__ Cadd,
    int M, int N, int K,
    long long strideA, long long strideB, long long strideC,
    float alpha, int bIsKN, int bF32, int bmode)
{
  __shared__ u16 As[2][TM * LDT];
  __shared__ u16 Bs[2][TN * LDT];

  const int z = blockIdx.z;
  long long offB;
  if (bmode == 1) {
    int g4 = (z < 16) ? (4 + (z >> 2)) : ((z - 16) >> 2);
    offB = (long long)g4 * (4LL * 192 * 2048);
  } else {
    offB = (long long)z * strideB;
  }
  const u16*   Bb16 = (const u16*)Bm + offB;
  const float* Bb32 = (const float*)Bm + offB;
  const u16*   Ab   = A + (long long)z * strideA;

  const int tid  = threadIdx.x;
  const int lane = tid & 31;
  const int wave = tid >> 5;
  const int m0   = (wave >> 1) * 16;
  const int n0   = (wave & 1) * 32;
  const int mblk = blockIdx.y * TM;
  const int nblk = blockIdx.x * TN;

  v8f acc0 = {};
  v8f acc1 = {};

  const int ar  = tid >> 2;          // 0..63  (row for A/B [*,K] staging)
  const int acs = (tid & 3) * 8;     // 0,8,16,24 (k segment)
  const int bkr = tid >> 3;          // 0..31  (k row for [K,N] staging)
  const int bns = (tid & 7) * 8;     // n segment for [K,N] staging

  // block-uniform: whole tile in range -> async DMA staging is safe
  const bool fullA = (mblk + TM <= M);
  const bool fullB = (!bIsKN) && (!bF32) && (nblk + TN <= N);

  auto stageA = [&](int kk, int buf) {
    const int gr = mblk + ar;
    u16* dst = &As[buf][ar * LDT + acs];
    const u16* src = Ab + (long long)gr * K + (kk + acs);
    if (fullA) {
      async_ld16(dst, src);
    } else {
      u16 tmp[8];
      if (gr < M) {
        #pragma unroll
        for (int j = 0; j < 8; ++j) tmp[j] = src[j];
      } else {
        #pragma unroll
        for (int j = 0; j < 8; ++j) tmp[j] = 0;
      }
      #pragma unroll
      for (int j = 0; j < 8; ++j) dst[j] = tmp[j];
    }
  };

  auto stageB = [&](int kk, int buf) {
    if (!bIsKN) {
      const int gn = nblk + ar;
      u16* dst = &Bs[buf][ar * LDT + acs];
      if (fullB) {
        async_ld16(dst, Bb16 + (long long)gn * K + (kk + acs));
      } else {
        u16 tmp[8];
        if (gn < N) {
          if (bF32) {
            const float* src = Bb32 + (long long)gn * K + (kk + acs);
            #pragma unroll
            for (int j = 0; j < 8; ++j) tmp[j] = f2bf(src[j]);
          } else {
            const u16* src = Bb16 + (long long)gn * K + (kk + acs);
            #pragma unroll
            for (int j = 0; j < 8; ++j) tmp[j] = src[j];
          }
        } else {
          #pragma unroll
          for (int j = 0; j < 8; ++j) tmp[j] = 0;
        }
        #pragma unroll
        for (int j = 0; j < 8; ++j) dst[j] = tmp[j];
      }
    } else {
      const long long krow = (long long)(kk + bkr) * N;
      if (kk + TK < K) {
        if (bF32) __builtin_prefetch((const void*)(Bb32 + krow + (long long)TK * N + nblk + bns), 0, 1);
        else      __builtin_prefetch((const void*)(Bb16 + krow + (long long)TK * N + nblk + bns), 0, 1);
      }
      #pragma unroll
      for (int j = 0; j < 8; ++j) {
        const int gn = nblk + bns + j;
        u16 v = 0;
        if (gn < N) v = bF32 ? f2bf(Bb32[krow + gn]) : Bb16[krow + gn];
        Bs[buf][(bns + j) * LDT + bkr] = v;
      }
    }
  };

  const int nk = K / TK;   // all K in this problem are multiples of 32

  stageA(0, 0);
  stageB(0, 0);
  wait_async();
  __syncthreads();

  const int kb = (lane & 16) ? 8 : 0;   // lanes 16-31 carry K {8..15,24..31}
  const int arow = (m0 + (lane & 15)) * LDT + kb;
  const int brow0 = (n0 + (lane & 15)) * LDT + kb;
  const int brow1 = (n0 + 16 + (lane & 15)) * LDT + kb;

  for (int t = 0; t < nk; ++t) {
    const int cur = t & 1;
    if (t + 1 < nk) {          // DMA next tile into the other buffer
      stageA((t + 1) * TK, cur ^ 1);
      stageB((t + 1) * TK, cur ^ 1);
    }
    FragBF fa, fb0, fb1;
    {
      const u16* p = &As[cur][arow];
      #pragma unroll
      for (int j = 0; j < 8; ++j) { fa.u[j] = p[j]; fa.u[8 + j] = p[16 + j]; }
    }
    {
      const u16* p = &Bs[cur][brow0];
      #pragma unroll
      for (int j = 0; j < 8; ++j) { fb0.u[j] = p[j]; fb0.u[8 + j] = p[16 + j]; }
      const u16* q = &Bs[cur][brow1];
      #pragma unroll
      for (int j = 0; j < 8; ++j) { fb1.u[j] = q[j]; fb1.u[8 + j] = q[16 + j]; }
    }
    acc0 = __builtin_amdgcn_wmma_f32_16x16x32_bf16(false, fa.v, false, fb0.v, (short)0, acc0, false, false);
    acc1 = __builtin_amdgcn_wmma_f32_16x16x32_bf16(false, fa.v, false, fb1.v, (short)0, acc1, false, false);
    wait_async();
    __syncthreads();
  }

  // ---- store per 16x16 f32 C/D layout ----
  const long long cb = (long long)z * strideC;
  #pragma unroll
  for (int i = 0; i < 8; ++i) {
    const int gr = mblk + m0 + i + ((lane >> 4) << 3);
    if (gr < M) {
      const int gc0 = nblk + n0 + (lane & 15);
      if (gc0 < N) {
        const long long idx = cb + (long long)gr * N + gc0;
        float v = alpha * acc0[i] + (Cadd ? Cadd[idx] : 0.0f);
        if (C)   C[idx]   = v;
        if (Cbf) Cbf[idx] = f2bf(v);
      }
      const int gc1 = nblk + n0 + 16 + (lane & 15);
      if (gc1 < N) {
        const long long idx = cb + (long long)gr * N + gc1;
        float v = alpha * acc1[i] + (Cadd ? Cadd[idx] : 0.0f);
        if (C)   C[idx]   = v;
        if (Cbf) Cbf[idx] = f2bf(v);
      }
    }
  }
}

// ---------------------------------------------------------------------------
// Auxiliary kernels
// ---------------------------------------------------------------------------

// xmean[b,c] = mean over N=192 of x[b,c,:]
__global__ void kmean_x(const float* __restrict__ x, float* __restrict__ xmean) {
  int idx = blockIdx.x * 256 + threadIdx.x;
  if (idx >= 32 * 2048) return;
  const float* p = x + (long long)idx * 192;
  float s = 0.f;
  for (int n = 0; n < 192; ++n) s += p[n];
  xmean[idx] = s * (1.0f / 192.0f);
}

// norms[b,n] = max(||x[b,:,n]||, 1e-12)
__global__ void knorm_x(const float* __restrict__ x, float* __restrict__ norms) {
  __shared__ float red[256];
  const int bn = blockIdx.x, b = bn / 192, n = bn % 192;
  const float* base = x + (long long)b * 2048 * 192 + n;
  float s = 0.f;
  for (int c = threadIdx.x; c < 2048; c += 256) { float v = base[(long long)c * 192]; s += v * v; }
  red[threadIdx.x] = s; __syncthreads();
  for (int st = 128; st > 0; st >>= 1) { if (threadIdx.x < st) red[threadIdx.x] += red[threadIdx.x + st]; __syncthreads(); }
  if (threadIdx.x == 0) norms[bn] = fmaxf(sqrtf(red[0]), 1e-12f);
}

// xt_bf[b,n,c]=bf16(x[b,c,n]); gn_bf = normalized copy
__global__ void ktranspose(const float* __restrict__ x, const float* __restrict__ norms,
                           u16* __restrict__ xt, u16* __restrict__ gn) {
  const int bn = blockIdx.x, b = bn / 192, n = bn % 192;
  const float inv = 1.0f / norms[bn];
  const float* base = x + (long long)b * 2048 * 192 + n;
  u16* xo = xt + (long long)bn * 2048;
  u16* go = gn + (long long)bn * 2048;
  for (int c = threadIdx.x; c < 2048; c += 256) {
    float v = base[(long long)c * 192];
    xo[c] = f2bf(v); go[c] = f2bf(v * inv);
  }
}

// row softmax, f32 in -> bf16 out
__global__ void ksoftmax(const float* __restrict__ in, u16* __restrict__ out, int L) {
  __shared__ float red[256];
  const long long row = blockIdx.x;
  const float* xr = in + row * L;
  u16* orow = out + row * L;
  const int tid = threadIdx.x;
  float m = -1e30f;
  for (int i = tid; i < L; i += 256) m = fmaxf(m, xr[i]);
  red[tid] = m; __syncthreads();
  for (int st = 128; st > 0; st >>= 1) { if (tid < st) red[tid] = fmaxf(red[tid], red[tid + st]); __syncthreads(); }
  m = red[0]; __syncthreads();
  float s = 0.f;
  for (int i = tid; i < L; i += 256) s += __expf(xr[i] - m);
  red[tid] = s; __syncthreads();
  for (int st = 128; st > 0; st >>= 1) { if (tid < st) red[tid] += red[tid + st]; __syncthreads(); }
  const float inv = 1.0f / red[0];
  for (int i = tid; i < L; i += 256) orow[i] = f2bf(__expf(xr[i] - m) * inv);
}

// q_bf[b,i,:] = bf16(query(b<16?v:i)[i,:] + xmean[b,:])
__global__ void kprep_q(const float* __restrict__ qv, const float* __restrict__ qi,
                        const float* __restrict__ xmean, u16* __restrict__ out) {
  const int row = blockIdx.x, b = row >> 4, i = row & 15;
  const float* qs = (b < 16 ? qv : qi) + (long long)i * 2048;
  const float* xm = xmean + (long long)b * 2048;
  u16* o = out + (long long)row * 2048;
  for (int c = threadIdx.x; c < 2048; c += 256) o[c] = f2bf(qs[c] + xm[c]);
}

// f_rel row 0 = xmean
__global__ void krow0(const float* __restrict__ xmean, float* __restrict__ dout0) {
  int idx = blockIdx.x * 256 + threadIdx.x;
  if (idx >= 32 * 2048) return;
  const int b = idx >> 11, c = idx & 2047;
  dout0[(long long)b * 17 * 2048 + c] = xmean[idx];
}

// frmean[b,c] = mean over 17 rows of f_rel[b]
__global__ void kmean17(const float* __restrict__ f, float* __restrict__ out) {
  int idx = blockIdx.x * 256 + threadIdx.x;
  if (idx >= 32 * 2048) return;
  const int b = idx >> 11, c = idx & 2047;
  const float* p = f + (long long)b * 17 * 2048 + c;
  float s = 0.f;
  for (int q = 0; q < 17; ++q) s += p[(long long)q * 2048];
  out[idx] = s * (1.0f / 17.0f);
}

// layernorm of f_rel rows (v/i params by batch half) -> f32 + bf16
__global__ void kln_q(const float* __restrict__ frel,
                      const float* wv, const float* bv, const float* wi, const float* bi,
                      float* __restrict__ outf, u16* __restrict__ outb) {
  __shared__ float r1[256]; __shared__ float r2[256];
  const int row = blockIdx.x, b = row / 17, tid = threadIdx.x;
  const float* w  = (b < 16) ? wv : wi;
  const float* bb = (b < 16) ? bv : bi;
  const float* xr = frel + (long long)row * 2048;
  float s = 0.f, s2 = 0.f;
  for (int c = tid; c < 2048; c += 256) { float v = xr[c]; s += v; s2 += v * v; }
  r1[tid] = s; r2[tid] = s2; __syncthreads();
  for (int st = 128; st > 0; st >>= 1) { if (tid < st) { r1[tid] += r1[tid + st]; r2[tid] += r2[tid + st]; } __syncthreads(); }
  const float mu = r1[0] * (1.0f / 2048.0f);
  const float var = r2[0] * (1.0f / 2048.0f) - mu * mu;
  const float rstd = rsqrtf(var + 1e-5f);
  for (int c = tid; c < 2048; c += 256) {
    float v = (xr[c] - mu) * rstd * w[c] + bb[c];
    outf[(long long)row * 2048 + c] = v;
    outb[(long long)row * 2048 + c] = f2bf(v);
  }
}

// layernorm of (prototype - frmean) rows, chunk of 8 batches -> bf16
__global__ void kln_g(const float* __restrict__ proto, const float* __restrict__ frmean,
                      const float* wv, const float* bv, const float* wi, const float* bi,
                      u16* __restrict__ out, int b0) {
  __shared__ float r1[256]; __shared__ float r2[256];
  const int lrow = blockIdx.x, tid = threadIdx.x;
  const int b = b0 + (lrow >> 10), p = lrow & 1023;
  const float* w  = (b < 16) ? wv : wi;
  const float* bb = (b < 16) ? bv : bi;
  const float* pr = proto + (long long)p * 2048;
  const float* fm = frmean + (long long)b * 2048;
  float s = 0.f, s2 = 0.f;
  for (int c = tid; c < 2048; c += 256) { float v = pr[c] - fm[c]; s += v; s2 += v * v; }
  r1[tid] = s; r2[tid] = s2; __syncthreads();
  for (int st = 128; st > 0; st >>= 1) { if (tid < st) { r1[tid] += r1[tid + st]; r2[tid] += r2[tid + st]; } __syncthreads(); }
  const float mu = r1[0] * (1.0f / 2048.0f);
  const float var = r2[0] * (1.0f / 2048.0f) - mu * mu;
  const float rstd = rsqrtf(var + 1e-5f);
  for (int c = tid; c < 2048; c += 256)
    out[(long long)lrow * 2048 + c] = f2bf(((pr[c] - fm[c]) - mu) * rstd * w[c] + bb[c]);
}

// q for f_cor (mode 0) / f_rec (mode 1): bf16(f_pro + mean(gathered g))
__global__ void kprep_q2(const float* __restrict__ fpro, const float* __restrict__ xmean,
                         u16* __restrict__ out, int mode) {
  const int row = blockIdx.x, b = row / 17;
  int s0, s1, s2, s3;
  if (mode == 0) {
    const int g4 = (b < 16) ? (4 + (b >> 2)) : ((b - 16) >> 2);
    s0 = 4 * g4; s1 = s0 + 1; s2 = s0 + 2; s3 = s0 + 3;
  } else {
    s0 = b;
    s1 = (b < 16) ? ((b + 4) & 15) : (16 + ((b - 12) & 15));
    s2 = s0; s3 = s1;
  }
  const float* m0 = xmean + (long long)s0 * 2048;
  const float* m1 = xmean + (long long)s1 * 2048;
  const float* m2 = xmean + (long long)s2 * 2048;
  const float* m3 = xmean + (long long)s3 * 2048;
  const float* fp = fpro + (long long)row * 2048;
  u16* o = out + (long long)row * 2048;
  for (int c = threadIdx.x; c < 2048; c += 256)
    o[c] = f2bf(fp[c] + 0.25f * (m0[c] + m1[c] + m2[c] + m3[c]));
}

// gather gws windows into x2 layout [32][2][192][2048] bf16
__global__ void kgather(const u16* __restrict__ gws, u16* __restrict__ g2) {
  const int win = blockIdx.x, b = win >> 1, w = win & 1;
  const int src = (w == 0) ? b : ((b < 16) ? ((b + 4) & 15) : (16 + ((b - 12) & 15)));
  const uint4* s = (const uint4*)(gws + (long long)src * 192 * 2048);
  uint4* d = (uint4*)(g2 + (long long)win * 192 * 2048);
  for (int i = threadIdx.x; i < 49152; i += 256) d[i] = s[i];
}

// ---------------------------------------------------------------------------
extern "C" void kernel_launch(void* const* d_in, const int* in_sizes, int n_in,
                              void* d_out, int out_size, void* d_ws, size_t ws_size,
                              hipStream_t stream)
{
  (void)in_sizes; (void)n_in; (void)out_size; (void)ws_size;
  const float* x      = (const float*)d_in[0];
  const float* qv     = (const float*)d_in[2];
  const float* qi     = (const float*)d_in[3];
  const float* proto  = (const float*)d_in[4];
  const float* q2r_wq = (const float*)d_in[5];
  const float* q2r_wg = (const float*)d_in[6];
  const float* p2r_wq = (const float*)d_in[7];
  const float* p2r_wg = (const float*)d_in[8];
  const float* v_wq   = (const float*)d_in[9];
  const float* v_wg   = (const float*)d_in[10];
  const float* i_wq   = (const float*)d_in[11];
  const float* i_wg   = (const float*)d_in[12];
  const float* v_lnx_w = (const float*)d_in[13];
  const float* v_lnx_b = (const float*)d_in[14];
  const float* v_lng_w = (const float*)d_in[15];
  const float* v_lng_b = (const float*)d_in[16];
  const float* i_lnx_w = (const float*)d_in[17];
  const float* i_lnx_b = (const float*)d_in[18];
  const float* i_lng_w = (const float*)d_in[19];
  const float* i_lng_b = (const float*)d_in[20];

  float* dout0 = (float*)d_out;                  // f_rel
  const long long S = 32LL * 17 * 2048;
  float* dout1 = dout0 + S;                      // f_pro
  float* dout2 = dout0 + 2 * S;                  // f_rec
  float* dout3 = dout0 + 3 * S;                  // f_cor

  char* wp = (char*)d_ws;
  auto carve = [&](size_t bytes) -> char* {
    char* p = wp; wp += (bytes + 255) & ~(size_t)255; return p;
  };
  u16* xt_bf  = (u16*)carve(32LL * 192 * 2048 * 2);
  u16* gn_bf  = (u16*)carve(32LL * 192 * 2048 * 2);
  u16* gws_bf = (u16*)carve(32LL * 192 * 2048 * 2);
  u16* g2_bf  = (u16*)carve(64LL * 192 * 2048 * 2);
  u16* gln_bf = (u16*)carve(8LL * 1024 * 2048 * 2);
  float* norms  = (float*)carve(32LL * 192 * 4);
  float* xmean  = (float*)carve(32LL * 2048 * 4);
  float* frmean = (float*)carve(32LL * 2048 * 4);
  float* relw = (float*)carve(32LL * 192 * 192 * 4);
  u16*   smw  = (u16*)carve(32LL * 192 * 192 * 2);
  u16* qbuf  = (u16*)carve(512LL * 2048 * 2);
  u16* relq1 = (u16*)carve(512LL * 2048 * 2);
  u16* rq2   = (u16*)carve(512LL * 2048 * 2);
  float* log1 = (float*)carve(32LL * 16 * 192 * 4);
  u16*   sm1  = (u16*)carve(32LL * 16 * 192 * 2);
  float* qlnf = (float*)carve(544LL * 2048 * 4);
  u16*   qlnb = (u16*)carve(544LL * 2048 * 2);
  u16* relqd = (u16*)carve(544LL * 2048 * 2);
  u16* qWb   = (u16*)carve(544LL * 2048 * 2);
  float* log2 = (float*)carve(544LL * 1024 * 4);
  u16*   sm2  = (u16*)carve(544LL * 1024 * 2);
  u16* qcor  = (u16*)carve(544LL * 2048 * 2);
  u16* qrec  = (u16*)carve(544LL * 2048 * 2);
  u16* relqc = (u16*)carve(544LL * 2048 * 2);
  u16* rq2c  = (u16*)carve(544LL * 2048 * 2);
  u16* relqr = (u16*)carve(544LL * 2048 * 2);
  u16* rq2r  = (u16*)carve(544LL * 2048 * 2);
  float* logc = (float*)carve(544LL * 768 * 4);
  u16*   smc  = (u16*)carve(544LL * 768 * 2);
  float* logr = (float*)carve(544LL * 384 * 4);
  u16*   smr  = (u16*)carve(544LL * 384 * 2);

  auto gemm = [&](const u16* A, const void* B, float* C, u16* Cbf, const float* Cadd,
                  int M, int N, int K, long long sA, long long sB, long long sC,
                  float alpha, int bIsKN, int bF32, int bmode, int batch) {
    dim3 g((N + 63) / 64, (M + 63) / 64, batch);
    gemm_bf16<<<g, 256, 0, stream>>>(A, B, C, Cbf, Cadd, M, N, K, sA, sB, sC,
                                     alpha, bIsKN, bF32, bmode);
  };

  // ---- prep: transpose x, means, norms ----
  kmean_x   <<<256, 256, 0, stream>>>(x, xmean);
  knorm_x   <<<6144, 256, 0, stream>>>(x, norms);
  ktranspose<<<6144, 256, 0, stream>>>(x, norms, xt_bf, gn_bf);

  // ---- window attention, once per source batch: gws = softmax(gn gn^T * s) @ xt ----
  gemm(gn_bf, gn_bf, relw, nullptr, nullptr, 192, 192, 2048,
       192LL * 2048, 192LL * 2048, 192LL * 192, SCALE_F, 0, 0, 0, 32);
  ksoftmax<<<32 * 192, 256, 0, stream>>>(relw, smw, 192);
  gemm(smw, xt_bf, nullptr, gws_bf, nullptr, 192, 2048, 192,
       192LL * 192, 192LL * 2048, 192LL * 2048, 1.0f, 1, 0, 0, 32);

  // ---- f_rel: logits = ((q Wq^T) Wg) gws^T ----
  kprep_q<<<512, 256, 0, stream>>>(qv, qi, xmean, qbuf);
  gemm(qbuf,  q2r_wq, nullptr, relq1, nullptr, 16, 2048, 2048,
       16LL * 2048, 0, 16LL * 2048, 1.0f, 0, 1, 0, 32);
  gemm(relq1, q2r_wg, nullptr, rq2,   nullptr, 16, 2048, 2048,
       16LL * 2048, 0, 16LL * 2048, 1.0f, 1, 1, 0, 32);
  gemm(rq2, gws_bf, log1, nullptr, nullptr, 16, 192, 2048,
       16LL * 2048, 192LL * 2048, 16LL * 192, SCALE_F, 0, 0, 0, 32);
  ksoftmax<<<512, 256, 0, stream>>>(log1, sm1, 192);
  gemm(sm1, gws_bf, dout0 + 2048, nullptr, nullptr, 16, 2048, 192,
       16LL * 192, 192LL * 2048, 17LL * 2048, 1.0f, 1, 0, 0, 32);
  krow0<<<256, 256, 0, stream>>>(xmean, dout0);

  // ---- f_pro (_deltaor), chunked over batches ----
  kmean17<<<256, 256, 0, stream>>>(dout0, frmean);
  kln_q<<<544, 256, 0, stream>>>(dout0, v_lnx_w, v_lnx_b, i_lnx_w, i_lnx_b, qlnf, qlnb);
  const long long HO = 16LL * 17 * 2048;
  gemm(qlnb,      v_wq, nullptr, relqd,      nullptr, 17, 2048, 2048,
       17LL * 2048, 0, 17LL * 2048, 1.0f, 0, 1, 0, 16);
  gemm(qlnb + HO, i_wq, nullptr, relqd + HO, nullptr, 17, 2048, 2048,
       17LL * 2048, 0, 17LL * 2048, 1.0f, 0, 1, 0, 16);
  gemm(relqd,      v_wg, nullptr, qWb,      nullptr, 17, 2048, 2048,
       17LL * 2048, 0, 17LL * 2048, 1.0f, 1, 1, 0, 16);
  gemm(relqd + HO, i_wg, nullptr, qWb + HO, nullptr, 17, 2048, 2048,
       17LL * 2048, 0, 17LL * 2048, 1.0f, 1, 1, 0, 16);
  for (int ch = 0; ch < 4; ++ch) {
    const int b0 = ch * 8;
    const long long qo = (long long)b0 * 17 * 2048;
    const long long lo = (long long)b0 * 17 * 1024;
    kln_g<<<8192, 256, 0, stream>>>(proto, frmean, v_lng_w, v_lng_b, i_lng_w, i_lng_b, gln_bf, b0);
    gemm(qWb + qo, gln_bf, log2 + lo, nullptr, nullptr, 17, 1024, 2048,
         17LL * 2048, 1024LL * 2048, 17LL * 1024, SCALE_F, 0, 0, 0, 8);
    ksoftmax<<<8 * 17, 256, 0, stream>>>(log2 + lo, sm2 + lo, 1024);
    gemm(sm2 + lo, gln_bf, dout1 + qo, nullptr, qlnf + qo, 17, 2048, 1024,
         17LL * 1024, 1024LL * 2048, 17LL * 2048, 1.0f, 1, 0, 0, 8);
  }

  // ---- f_cor / f_rec catchers on f_pro ----
  kprep_q2<<<544, 256, 0, stream>>>(dout1, xmean, qcor, 0);
  kprep_q2<<<544, 256, 0, stream>>>(dout1, xmean, qrec, 1);
  gemm(qcor,  p2r_wq, nullptr, relqc, nullptr, 17, 2048, 2048,
       17LL * 2048, 0, 17LL * 2048, 1.0f, 0, 1, 0, 32);
  gemm(relqc, p2r_wg, nullptr, rq2c,  nullptr, 17, 2048, 2048,
       17LL * 2048, 0, 17LL * 2048, 1.0f, 1, 1, 0, 32);
  gemm(qrec,  p2r_wq, nullptr, relqr, nullptr, 17, 2048, 2048,
       17LL * 2048, 0, 17LL * 2048, 1.0f, 0, 1, 0, 32);
  gemm(relqr, p2r_wg, nullptr, rq2r,  nullptr, 17, 2048, 2048,
       17LL * 2048, 0, 17LL * 2048, 1.0f, 1, 1, 0, 32);
  kgather<<<64, 256, 0, stream>>>(gws_bf, g2_bf);

  // f_cor: windows are 4 contiguous gws batches (bmode=1 offset mapping)
  gemm(rq2c, gws_bf, logc, nullptr, nullptr, 17, 768, 2048,
       17LL * 2048, 0, 17LL * 768, SCALE_F, 0, 0, 1, 32);
  ksoftmax<<<544, 256, 0, stream>>>(logc, smc, 768);
  gemm(smc, gws_bf, dout3, nullptr, nullptr, 17, 2048, 768,
       17LL * 768, 0, 17LL * 2048, 1.0f, 1, 0, 1, 32);

  // f_rec: gathered 2-window tensor
  gemm(rq2r, g2_bf, logr, nullptr, nullptr, 17, 384, 2048,
       17LL * 2048, 384LL * 2048, 17LL * 384, SCALE_F, 0, 0, 0, 32);
  ksoftmax<<<544, 256, 0, stream>>>(logr, smr, 384);
  gemm(smr, g2_bf, dout2, nullptr, nullptr, 17, 2048, 384,
       17LL * 384, 384LL * 2048, 17LL * 2048, 1.0f, 1, 0, 0, 32);
}
